// AE_Transformer_47339129536748
// MI455X (gfx1250) — compile-verified, matrix operand order checked by reading, not run on hardware
//
#include <hip/hip_runtime.h>

// ---------------------------------------------------------------------------
// AE-Transformer forward for MI455X (gfx1250, wave32).
// All 57 linear layers -> padded-bf16 staging + double-buffered async-LDS
// WMMA GEMM:
//   global_load_async_to_lds_b128 (ASYNCcnt, in-order completion)
//   -> ping-pong LDS buffers with s_wait_asynccnt 8 partial waits
//   -> ds_load_b128 fragments -> v_wmma_f32_16x16x32_bf16 (f32 accumulate).
// ---------------------------------------------------------------------------

typedef __attribute__((ext_vector_type(16))) __bf16 bf16x16;
typedef __attribute__((ext_vector_type(8)))  float  floatx8;

union FragU { bf16x16 v; uint4 q[2]; };

__device__ __forceinline__ unsigned short f2bf(float x) {
    unsigned u = __float_as_uint(x);
    unsigned r = (u + 0x7FFFu + ((u >> 16) & 1u)) >> 16;   // round-to-nearest-even
    return (unsigned short)r;
}

// low 32 bits of a generic pointer to LDS == workgroup-relative LDS byte addr
__device__ __forceinline__ unsigned lds_addr32(const void* p) {
    return (unsigned)(uintptr_t)p;
}

// CDNA5 async copy global -> LDS (16B per lane), tracked with ASYNCcnt.
__device__ __forceinline__ void async_ld_b128(unsigned lds_off, const void* gaddr) {
    asm volatile("global_load_async_to_lds_b128 %0, %1, off"
                 :: "v"(lds_off), "v"((unsigned long long)(uintptr_t)gaddr)
                 : "memory");
}
// wait all async copies done
__device__ __forceinline__ void wait_async0() {
    asm volatile("s_wait_asynccnt 0x0" ::: "memory");
}
// wait until <= 8 outstanding: with 8 issues per tile and in-order completion
// this means "previous tile fully landed, next tile still in flight".
__device__ __forceinline__ void wait_async_le8() {
    asm volatile("s_wait_asynccnt 0x8" ::: "memory");
}

// ---------------------------------------------------------------------------
// fp32 -> bf16 with zero padding: dst is (Rp x Kp), src is (R x K) contiguous.
// Padding rows/cols are zero-filled so the GEMM needs no load guards.
// ---------------------------------------------------------------------------
__global__ __launch_bounds__(256)
void cvt_bf16_pad_kernel(const float* __restrict__ src, unsigned short* __restrict__ dst,
                         int R, int K, int Kp, int total) {
    int idx = blockIdx.x * 256 + threadIdx.x;
    if (idx >= total) return;
    int k = idx % Kp;
    int r = idx / Kp;
    float v = (r < R && k < K) ? src[(size_t)r * K + k] : 0.f;
    dst[idx] = f2bf(v);
}

// ---------------------------------------------------------------------------
// GEMM: C[M,N] = act(A[Mp,Kp]_bf16 * W[Np,Kp]_bf16^T + bias[N])
// Block 128x128x64, 8 waves (2x4), wave tile 64x32 = 4x2 WMMA fragments.
// Double-buffered LDS staged with async b128 copies; 2 WMMA K-steps / tile.
// ---------------------------------------------------------------------------
#define LDSROW 72   // 64 bf16 + 8 pad -> 144B stride, conflict-free b128 reads

__global__ __launch_bounds__(256)
void wgemm_bf16_kernel(const unsigned short* __restrict__ A,
                       const unsigned short* __restrict__ W,
                       int Kp, const float* __restrict__ bias,
                       float* __restrict__ C, int ldc,
                       int M, int N, int act) {
    __shared__ __align__(16) unsigned short As[2][128 * LDSROW];
    __shared__ __align__(16) unsigned short Bs[2][128 * LDSROW];

    const int tid  = threadIdx.x;
    const int lane = tid & 31;
    const int wid  = tid >> 5;
    const int wm   = (wid & 1) * 64;
    const int wn   = (wid >> 1) * 32;
    const size_t bM = (size_t)blockIdx.y * 128;
    const size_t bN = (size_t)blockIdx.x * 128;

    floatx8 acc[4][2] = {};

    // staging map: 1024 16B-chunks per tile, 4 consecutive chunks per thread
    // chunk c -> row c>>3, col ((c&7)*8); c0 % 8 in {0,4} so row is constant.
    const int c0   = tid * 4;
    const int srow = c0 >> 3;
    const int scol = (c0 & 7) * 8;                  // 0 or 32 (bf16 units)
    const unsigned lA0 = lds_addr32(&As[0][srow * LDSROW + scol]);
    const unsigned lA1 = lds_addr32(&As[1][srow * LDSROW + scol]);
    const unsigned lB0 = lds_addr32(&Bs[0][srow * LDSROW + scol]);
    const unsigned lB1 = lds_addr32(&Bs[1][srow * LDSROW + scol]);
    const unsigned short* ga = A + (bM + srow) * (size_t)Kp + scol;
    const unsigned short* gb = W + (bN + srow) * (size_t)Kp + scol;

    auto stage = [&](int tile, int buf) {
        const int kb = tile * 64;
        const unsigned la = buf ? lA1 : lA0;
        const unsigned lb = buf ? lB1 : lB0;
#pragma unroll
        for (int c = 0; c < 4; ++c)
            async_ld_b128(la + c * 16, ga + kb + c * 8);
#pragma unroll
        for (int c = 0; c < 4; ++c)
            async_ld_b128(lb + c * 16, gb + kb + c * 8);
    };

    const int nt = Kp >> 6;     // number of 64-wide K tiles (Kp % 64 == 0)
    stage(0, 0);                // prologue: fill buffer 0

    for (int t = 0; t < nt; ++t) {
        const int cur = t & 1;
        if (t + 1 < nt) {
            stage(t + 1, cur ^ 1);   // keep next tile in flight ...
            wait_async_le8();        // ... while requiring current tile landed
        } else {
            wait_async0();
        }
        __syncthreads();

        const unsigned short* as = As[cur];
        const unsigned short* bs = Bs[cur];
#pragma unroll
        for (int ks = 0; ks < 2; ++ks) {
            // ISA 16-bit A-layout: lanes 0-15 K 0..7/16..23, lanes 16-31 +8
            const int kh = ks * 32 + ((lane >> 4) << 3);
            FragU a[4];
#pragma unroll
            for (int i = 0; i < 4; ++i) {
                int r = wm + i * 16 + (lane & 15);
                a[i].q[0] = *(const uint4*)&as[r * LDSROW + kh];
                a[i].q[1] = *(const uint4*)&as[r * LDSROW + kh + 16];
            }
#pragma unroll
            for (int j = 0; j < 2; ++j) {
                FragU b;
                int r = wn + j * 16 + (lane & 15);
                b.q[0] = *(const uint4*)&bs[r * LDSROW + kh];
                b.q[1] = *(const uint4*)&bs[r * LDSROW + kh + 16];
#pragma unroll
                for (int i = 0; i < 4; ++i)
                    acc[i][j] = __builtin_amdgcn_wmma_f32_16x16x32_bf16(
                        false, a[i].v, false, b.v, (short)0, acc[i][j], false, false);
            }
        }
        // all waves done reading buffer `cur` before anyone re-stages it
        __syncthreads();
    }

    // epilogue: D layout VGPR r -> M = r + 8*(lane>>4), N = lane&15
#pragma unroll
    for (int i = 0; i < 4; ++i) {
#pragma unroll
        for (int j = 0; j < 2; ++j) {
            int n = (int)bN + wn + j * 16 + (lane & 15);
            if (n >= N) continue;
            float bv = bias ? bias[n] : 0.f;
#pragma unroll
            for (int r = 0; r < 8; ++r) {
                int m = (int)bM + wm + i * 16 + r + ((lane >> 4) << 3);
                if (m < M) {
                    float v = acc[i][j][r] + bv;
                    if (act) v = fmaxf(v, 0.f);
                    C[(size_t)m * ldc + n] = v;
                }
            }
        }
    }
}

// ---------------------------------------------------------------------------
// Attention over time dim (T=8) per (patch n, head h). hd=64, tiny -> VALU.
// ---------------------------------------------------------------------------
__global__ __launch_bounds__(64)
void attn_kernel(const float* __restrict__ Q, const float* __restrict__ Km,
                 const float* __restrict__ Vm, int ld,
                 float* __restrict__ out, int NP, int causal) {
    const int n = blockIdx.x;
    const int h = threadIdx.x & 7;
    const int i = threadIdx.x >> 3;

    float q[64];
    const float* qp = Q + ((size_t)(i * NP + n)) * ld + h * 64;
#pragma unroll
    for (int d = 0; d < 64; ++d) q[d] = qp[d];

    float s[8];
    float mx = -1e30f;
#pragma unroll
    for (int j = 0; j < 8; ++j) {
        if (causal && j > i) { s[j] = -1e30f; continue; }
        const float* kp = Km + ((size_t)(j * NP + n)) * ld + h * 64;
        float acc = 0.f;
#pragma unroll
        for (int d = 0; d < 64; ++d) acc += q[d] * kp[d];
        s[j] = acc * 0.125f;
        mx = fmaxf(mx, s[j]);
    }
    float den = 0.f;
#pragma unroll
    for (int j = 0; j < 8; ++j) {
        s[j] = (causal && j > i) ? 0.f : __expf(s[j] - mx);
        den += s[j];
    }
    float inv = 1.f / den;
    float o[64];
#pragma unroll
    for (int d = 0; d < 64; ++d) o[d] = 0.f;
#pragma unroll
    for (int j = 0; j < 8; ++j) {
        float a = s[j] * inv;
        const float* vp = Vm + ((size_t)(j * NP + n)) * ld + h * 64;
#pragma unroll
        for (int d = 0; d < 64; ++d) o[d] += a * vp[d];
    }
    float* op = out + ((size_t)(i * NP + n)) * 512 + h * 64;
#pragma unroll
    for (int d = 0; d < 64; ++d) op[d] = o[d];
}

// ---------------------------------------------------------------------------
// Fused residual-add + LayerNorm over H=512; one wave32 per row.
// ---------------------------------------------------------------------------
__global__ __launch_bounds__(256)
void add_ln_kernel(const float* __restrict__ X, const float* __restrict__ R,
                   const float* __restrict__ S, const float* __restrict__ Bb,
                   float* __restrict__ O, int rows) {
    const int row  = blockIdx.x * 8 + (threadIdx.x >> 5);
    const int lane = threadIdx.x & 31;
    if (row >= rows) return;
    float v[16];
#pragma unroll
    for (int u = 0; u < 16; ++u) {
        int c = lane + u * 32;
        float t = X[(size_t)row * 512 + c];
        if (R) t += R[(size_t)row * 512 + c];
        v[u] = t;
    }
    float sum = 0.f;
#pragma unroll
    for (int u = 0; u < 16; ++u) sum += v[u];
#pragma unroll
    for (int m = 16; m > 0; m >>= 1) sum += __shfl_xor(sum, m, 32);
    const float mean = sum * (1.f / 512.f);
    float vs = 0.f;
#pragma unroll
    for (int u = 0; u < 16; ++u) { float d = v[u] - mean; vs += d * d; }
#pragma unroll
    for (int m = 16; m > 0; m >>= 1) vs += __shfl_xor(vs, m, 32);
    const float rstd = rsqrtf(vs * (1.f / 512.f) + 1e-5f);
#pragma unroll
    for (int u = 0; u < 16; ++u) {
        int c = lane + u * 32;
        O[(size_t)row * 512 + c] = (v[u] - mean) * rstd * S[c] + Bb[c];
    }
}

// ---------------------------------------------------------------------------
// Build (T=8, NP=257, 512): patch0 = 0, else feat[t*256+p-1]; plus sinusoidal
// positional encoding at pos = peOff + p, scaled by (1 + sqrt(512)).
// ---------------------------------------------------------------------------
__global__ void build_seq_kernel(const float* __restrict__ feat,
                                 float* __restrict__ dst, int peOff) {
    int idx = blockIdx.x * blockDim.x + threadIdx.x;
    const int total = 8 * 257 * 512;
    if (idx >= total) return;
    int c  = idx & 511;
    int rp = idx >> 9;
    int p  = rp % 257;
    int t  = rp / 257;
    float f = (p == 0) ? 0.f : feat[((size_t)t * 256 + (p - 1)) * 512 + c];
    float pos = (float)(peOff + p);
    float ang = pos * __powf(10000.f, -2.f * (float)c * (1.f / 512.f));
    float pe  = ((c & 1) ? __cosf(ang) : __sinf(ang)) * (1.f + 22.62741699796952f);
    dst[idx] = f + pe;
}

// drop patch 0: (8,257,512) -> (8,256,512)
__global__ void gather_kernel(const float* __restrict__ src, float* __restrict__ dst) {
    int idx = blockIdx.x * blockDim.x + threadIdx.x;
    const int total = 8 * 256 * 512;
    if (idx >= total) return;
    int c = idx & 511;
    int r = idx >> 9;
    int p = r % 256;
    int t = r / 256;
    dst[idx] = src[((size_t)t * 257 + p + 1) * 512 + c];
}

// deterministic two-stage MSE reduction
__global__ __launch_bounds__(256)
void sqdiff_partial_kernel(const float* __restrict__ a, const float* __restrict__ b,
                           float* __restrict__ partial, int ntot) {
    __shared__ float sm[256];
    float s = 0.f;
    for (int i = blockIdx.x * 256 + threadIdx.x; i < ntot; i += gridDim.x * 256) {
        float d = a[i] - b[i];
        s += d * d;
    }
    sm[threadIdx.x] = s;
    __syncthreads();
    for (int st = 128; st > 0; st >>= 1) {
        if (threadIdx.x < st) sm[threadIdx.x] += sm[threadIdx.x + st];
        __syncthreads();
    }
    if (threadIdx.x == 0) partial[blockIdx.x] = sm[0];
}

__global__ __launch_bounds__(256)
void loss_final_kernel(const float* __restrict__ partial, int nb,
                       float* __restrict__ out, float invN) {
    __shared__ float sm[256];
    float s = 0.f;
    for (int i = threadIdx.x; i < nb; i += 256) s += partial[i];
    sm[threadIdx.x] = s;
    __syncthreads();
    for (int st = 128; st > 0; st >>= 1) {
        if (threadIdx.x < st) sm[threadIdx.x] += sm[threadIdx.x + st];
        __syncthreads();
    }
    if (threadIdx.x == 0) { out[0] = sm[0] * invN; out[1] = 0.f; }
}

// ---------------------------------------------------------------------------
// Host-side orchestration (all on `stream`, scratch only from d_ws).
// ---------------------------------------------------------------------------
extern "C" void kernel_launch(void* const* d_in, const int* in_sizes, int n_in,
                              void* d_out, int out_size, void* d_ws, size_t ws_size,
                              hipStream_t stream) {
    (void)in_sizes; (void)n_in; (void)out_size; (void)ws_size;

    const float* x        = (const float*)d_in[0];
    const float* target   = (const float*)d_in[1];
    const float* lc_w     = (const float*)d_in[2];
    const float* lc_b     = (const float*)d_in[3];
    const float* ei_w     = (const float*)d_in[4];
    const float* ei_b     = (const float*)d_in[5];
    const float* eh_w     = (const float*)d_in[6];
    const float* eh_b     = (const float*)d_in[7];
    const float* eo_w     = (const float*)d_in[8];
    const float* eo_b     = (const float*)d_in[9];
    const float* di_w     = (const float*)d_in[10];
    const float* di_b     = (const float*)d_in[11];
    const float* dh_w     = (const float*)d_in[12];
    const float* dh_b     = (const float*)d_in[13];
    const float* do_w     = (const float*)d_in[14];
    const float* do_b     = (const float*)d_in[15];
    const float* te_qkv_w = (const float*)d_in[16];
    const float* te_qkv_b = (const float*)d_in[17];
    const float* te_o_w   = (const float*)d_in[18];
    const float* te_o_b   = (const float*)d_in[19];
    const float* te_ln1_s = (const float*)d_in[20];
    const float* te_ln1_b = (const float*)d_in[21];
    const float* te_f1_w  = (const float*)d_in[22];
    const float* te_f1_b  = (const float*)d_in[23];
    const float* te_f2_w  = (const float*)d_in[24];
    const float* te_f2_b  = (const float*)d_in[25];
    const float* te_ln2_s = (const float*)d_in[26];
    const float* te_ln2_b = (const float*)d_in[27];
    const float* te_lnf_s = (const float*)d_in[28];
    const float* te_lnf_b = (const float*)d_in[29];
    const float* sa_qkv_w = (const float*)d_in[30];
    const float* sa_qkv_b = (const float*)d_in[31];
    const float* sa_o_w   = (const float*)d_in[32];
    const float* sa_o_b   = (const float*)d_in[33];
    const float* td_ln1_s = (const float*)d_in[34];
    const float* td_ln1_b = (const float*)d_in[35];
    const float* ca_qkv_w = (const float*)d_in[36];
    const float* ca_qkv_b = (const float*)d_in[37];
    const float* ca_o_w   = (const float*)d_in[38];
    const float* ca_o_b   = (const float*)d_in[39];
    const float* td_ln2_s = (const float*)d_in[40];
    const float* td_ln2_b = (const float*)d_in[41];
    const float* td_f1_w  = (const float*)d_in[42];
    const float* td_f1_b  = (const float*)d_in[43];
    const float* td_f2_w  = (const float*)d_in[44];
    const float* td_f2_b  = (const float*)d_in[45];
    const float* td_ln3_s = (const float*)d_in[46];
    const float* td_ln3_b = (const float*)d_in[47];
    const float* td_lnf_s = (const float*)d_in[48];
    const float* td_lnf_b = (const float*)d_in[49];

    float* dec_out = (float*)d_out;   // (2048,2500) dec + loss + zero

    // --- workspace layout (fp32 units; all sizes multiples of 4 -> 16B align) ---
    float* ws = (float*)d_ws;
    size_t off = 0;
    auto alloc = [&](size_t n) { float* p = ws + off; off += n; return p; };
    float* lc_out = alloc((size_t)2048 * 2500);
    float* ffbuf  = alloc((size_t)2056 * 2048);
    float* h_enc  = alloc((size_t)2056 * 512);
    float* y_dec  = alloc((size_t)2056 * 512);
    float* membuf = alloc((size_t)2056 * 512);
    float* qkv    = alloc((size_t)2056 * 1536);
    float* attnb  = alloc((size_t)2056 * 512);
    float* tmp    = alloc((size_t)2056 * 512);
    float* src_f  = alloc((size_t)2048 * 512);
    float* tgt_f  = alloc((size_t)2048 * 512);
    float* h1     = alloc((size_t)2048 * 512);
    float* h2     = alloc((size_t)2048 * 512);
    float* out_c  = alloc((size_t)2048 * 512);
    float* part   = alloc(1024);
    // bf16 staging (padded): A up to 2048x7552, W up to 2560x7552
    unsigned short* Abf = (unsigned short*)alloc((size_t)2048 * 7552 / 2);
    unsigned short* Wbf = (unsigned short*)alloc((size_t)2560 * 7552 / 2);

    auto gemm = [&](const float* A, const float* W, const float* bias,
                    float* C, int ldc, int M, int N, int K, int act) {
        const int Kp = (K + 63) & ~63;
        const int Mp = (M + 127) & ~127;
        const int Np = (N + 127) & ~127;
        const int na = Mp * Kp, nw = Np * Kp;
        cvt_bf16_pad_kernel<<<(na + 255) / 256, 256, 0, stream>>>(A, Abf, M, K, Kp, na);
        cvt_bf16_pad_kernel<<<(nw + 255) / 256, 256, 0, stream>>>(W, Wbf, N, K, Kp, nw);
        dim3 g(Np / 128, Mp / 128);
        wgemm_bf16_kernel<<<g, 256, 0, stream>>>(Abf, Wbf, Kp, bias, C, ldc, M, N, act);
    };
    auto add_ln = [&](const float* X, const float* R, const float* S,
                      const float* Bb, float* O, int rows) {
        add_ln_kernel<<<(rows + 7) / 8, 256, 0, stream>>>(X, R, S, Bb, O, rows);
    };

    // ---- patch embeddings ----
    gemm(x, lc_w, lc_b, lc_out, 2500, 2048, 2500, 7500, 0);
    gemm(lc_out, ei_w, ei_b, h1, 512, 2048, 512, 2500, 0);
    gemm(h1, eh_w, eh_b, h2, 512, 2048, 512, 512, 1);
    gemm(h2, eh_w, eh_b, h1, 512, 2048, 512, 512, 1);
    gemm(h1, eo_w, eo_b, src_f, 512, 2048, 512, 512, 0);

    gemm(target, ei_w, ei_b, h1, 512, 2048, 512, 2500, 0);
    gemm(h1, eh_w, eh_b, h2, 512, 2048, 512, 512, 1);
    gemm(h2, eh_w, eh_b, h1, 512, 2048, 512, 512, 1);
    gemm(h1, eo_w, eo_b, tgt_f, 512, 2048, 512, 512, 0);

    // ---- concat zero row + positional encoding ----
    {
        int total = 8 * 257 * 512;
        build_seq_kernel<<<(total + 255) / 256, 256, 0, stream>>>(src_f, h_enc, 0);
        build_seq_kernel<<<(total + 255) / 256, 256, 0, stream>>>(tgt_f, y_dec, 257);
    }

    // ---- transformer encoder (4 layers) ----
    for (int l = 0; l < 4; ++l) {
        gemm(h_enc, te_qkv_w + (size_t)l * 1536 * 512, te_qkv_b + l * 1536,
             qkv, 1536, 2056, 1536, 512, 0);
        attn_kernel<<<257, 64, 0, stream>>>(qkv, qkv + 512, qkv + 1024, 1536, attnb, 257, 0);
        gemm(attnb, te_o_w + (size_t)l * 512 * 512, te_o_b + l * 512,
             tmp, 512, 2056, 512, 512, 0);
        add_ln(tmp, h_enc, te_ln1_s + l * 512, te_ln1_b + l * 512, h_enc, 2056);

        gemm(h_enc, te_f1_w + (size_t)l * 2048 * 512, te_f1_b + l * 2048,
             ffbuf, 2048, 2056, 2048, 512, 1);
        gemm(ffbuf, te_f2_w + (size_t)l * 512 * 2048, te_f2_b + l * 512,
             tmp, 512, 2056, 512, 2048, 0);
        add_ln(tmp, h_enc, te_ln2_s + l * 512, te_ln2_b + l * 512, h_enc, 2056);
    }
    add_ln(h_enc, nullptr, te_lnf_s, te_lnf_b, membuf, 2056);

    // ---- transformer decoder (4 layers) ----
    for (int l = 0; l < 4; ++l) {
        // self-attention (causal over time)
        gemm(y_dec, sa_qkv_w + (size_t)l * 1536 * 512, sa_qkv_b + l * 1536,
             qkv, 1536, 2056, 1536, 512, 0);
        attn_kernel<<<257, 64, 0, stream>>>(qkv, qkv + 512, qkv + 1024, 1536, attnb, 257, 1);
        gemm(attnb, sa_o_w + (size_t)l * 512 * 512, sa_o_b + l * 512,
             tmp, 512, 2056, 512, 512, 0);
        add_ln(tmp, y_dec, td_ln1_s + l * 512, td_ln1_b + l * 512, y_dec, 2056);

        // cross-attention: Q from y, K/V from encoder memory
        gemm(y_dec, ca_qkv_w + (size_t)l * 1536 * 512, ca_qkv_b + l * 1536,
             qkv, 1536, 2056, 512, 512, 0);
        gemm(membuf, ca_qkv_w + (size_t)l * 1536 * 512 + (size_t)512 * 512,
             ca_qkv_b + l * 1536 + 512, qkv + 512, 1536, 2056, 1024, 512, 0);
        attn_kernel<<<257, 64, 0, stream>>>(qkv, qkv + 512, qkv + 1024, 1536, attnb, 257, 0);
        gemm(attnb, ca_o_w + (size_t)l * 512 * 512, ca_o_b + l * 512,
             tmp, 512, 2056, 512, 512, 0);
        add_ln(tmp, y_dec, td_ln2_s + l * 512, td_ln2_b + l * 512, y_dec, 2056);

        // feed-forward
        gemm(y_dec, td_f1_w + (size_t)l * 2048 * 512, td_f1_b + l * 2048,
             ffbuf, 2048, 2056, 2048, 512, 1);
        gemm(ffbuf, td_f2_w + (size_t)l * 512 * 2048, td_f2_b + l * 512,
             tmp, 512, 2056, 512, 2048, 0);
        add_ln(tmp, y_dec, td_ln3_s + l * 512, td_ln3_b + l * 512, y_dec, 2056);
    }
    add_ln(y_dec, nullptr, td_lnf_s, td_lnf_b, tmp, 2056);   // final LN

    // drop patch 0 -> contiguous (2048, 512)
    {
        int total = 8 * 256 * 512;
        gather_kernel<<<(total + 255) / 256, 256, 0, stream>>>(tmp, out_c);
    }

    // ---- loss = mean((out - tgt_feat)^2) ----
    {
        const int ntot = 8 * 256 * 512;
        sqdiff_partial_kernel<<<512, 256, 0, stream>>>(out_c, tgt_f, part, ntot);
        loss_final_kernel<<<1, 256, 0, stream>>>(part, 512,
                                                 dec_out + (size_t)2048 * 2500,
                                                 1.f / (float)ntot);
    }

    // ---- output decoder MLP -> dec (2048, 2500) straight into d_out ----
    gemm(out_c, di_w, di_b, h1, 512, 2048, 512, 512, 0);
    gemm(h1, dh_w, dh_b, h2, 512, 2048, 512, 512, 1);
    gemm(h2, dh_w, dh_b, h1, 512, 2048, 512, 512, 1);
    gemm(h1, do_w, do_b, dec_out, 2500, 2048, 2500, 512, 0);
}
